// NeuralFractionalDE_10531259810488
// MI455X (gfx1250) — matrix operand order repaired and checked by reading, not compiled, not run on hardware
//
#include <hip/hip_runtime.h>
#include <math.h>

typedef __attribute__((ext_vector_type(16))) _Float16 v16h;
typedef __attribute__((ext_vector_type(8)))  float    v8f;
typedef __attribute__((ext_vector_type(4)))  float    v4f;

#define B_TOTAL 2048
#define D_DIM   128
#define HID_DIM 256
#define K_MEM   1024
#define BM      32          // batch rows per MLP block

#define DT_F       0.01f
#define ALPHA_DT_F 0.007f                  // ALPHA * DT
#define DT_POW_H_F 1.9952623149688797e-2f  // 0.01^0.85
#define INV_GAMMA  0.33427267776624133f    // 1 / Gamma(0.3)

__device__ __forceinline__ float fast_tanh(float x) {
#if __has_builtin(__builtin_amdgcn_tanhf)
  return __builtin_amdgcn_tanhf(x);        // native v_tanh_f32 on gfx1250
#else
  return tanhf(x);
#endif
}

// ---------------------------------------------------------------------------
// Telescoped Caputo coefficients: c[j] such that
// sum_k (x[k+1]-x[k]) * w[k]  ==  sum_j x[j] * c[j],   w[k] = (K-k)^(-a)/G
// ---------------------------------------------------------------------------
__global__ void coeff_kernel(float* __restrict__ c) {
  int j = blockIdx.x * blockDim.x + threadIdx.x;
  if (j >= K_MEM) return;
  float wj  = (j <= K_MEM - 2) ? powf((float)(K_MEM - j), -0.7f) * INV_GAMMA : 0.0f;
  float wjm = (j >= 1)         ? powf((float)(K_MEM - (j - 1)), -0.7f) * INV_GAMMA : 0.0f;
  float cj;
  if (j == 0)            cj = -wj;
  else if (j == K_MEM-1) cj = wjm;
  else                   cj = wjm - wj;
  c[j] = cj;
}

// ---------------------------------------------------------------------------
// Weight convert: fp32 -> f16, layout preserved ([K][N] row-major — this is
// the layout load_fragB reads: each lane gathers column n0+(lane&15),
// one half per k row, stride N, k = k0+8*(lane/16)+{0..7,16..23}).
// ---------------------------------------------------------------------------
__global__ void wconv_kernel(const float* __restrict__ src, _Float16* __restrict__ dst,
                             int total) {
  int i = blockIdx.x * blockDim.x + threadIdx.x;
  if (i >= total) return;
  dst[i] = (_Float16)src[i];
}

// ---------------------------------------------------------------------------
// Fractional derivative: fd[b][d] = sum_j x_history[b][j][d] * c[j]
// Single HBM pass over 1.07 GB (non-temporal). float4 per lane.
// ---------------------------------------------------------------------------
__global__ __launch_bounds__(256) void frac_kernel(const float* __restrict__ xh,
                                                   const float* __restrict__ c,
                                                   float* __restrict__ fd) {
  __shared__ float cs[K_MEM];
  for (int i = threadIdx.x; i < K_MEM; i += 256) cs[i] = c[i];
  __syncthreads();

  int row = threadIdx.x >> 5;          // 0..7 batch rows per block
  int d4  = threadIdx.x & 31;          // 0..31 -> 4 floats each, covers D=128
  int b   = blockIdx.x * 8 + row;

  const v4f* p = (const v4f*)(xh + (size_t)b * K_MEM * D_DIM) + d4;
  v4f acc = {0.f, 0.f, 0.f, 0.f};

  for (int k = 0; k < K_MEM; k += 8) {
#pragma unroll
    for (int u = 0; u < 8; ++u) {
      v4f v = __builtin_nontemporal_load(p + (size_t)(k + u) * (D_DIM / 4));
      float w = cs[k + u];
      acc.x = fmaf(v.x, w, acc.x);
      acc.y = fmaf(v.y, w, acc.y);
      acc.z = fmaf(v.z, w, acc.z);
      acc.w = fmaf(v.w, w, acc.w);
    }
  }
  ((v4f*)fd)[(size_t)b * (D_DIM / 4) + d4] = acc;
}

// ---------------------------------------------------------------------------
// WMMA A fragment (16-bit A 16x32, ISA 7.12.2): lane&15 selects row M;
// lanes 0-15 take K halves {k0..k0+7, k0+16..k0+23}, lanes 16-31 take
// {k0+8..k0+15, k0+24..k0+31}.
// ---------------------------------------------------------------------------
__device__ __forceinline__ v16h load_fragA(const _Float16* rowbase, int ld, int k0, int lane) {
  const _Float16* p = rowbase + (lane & 15) * ld + k0 + ((lane & 16) ? 8 : 0);
  union { v16h v; uint4 u[2]; } f;
  f.u[0] = *(const uint4*)(p);
  f.u[1] = *(const uint4*)(p + 16);
  return f.v;
}

// WMMA B fragment (16-bit B 32x16): lane&15 selects column N; lanes 0-15 take
// K {k0..k0+7, k0+16..k0+23}, lanes 16-31 take {k0+8..k0+15, k0+24..k0+31}
// (strided gather from a [K][N] row-major f16 weight).
__device__ __forceinline__ v16h load_fragB(const _Float16* W, int ldw, int k0, int n0, int lane) {
  const _Float16* p = W + (size_t)(k0 + ((lane & 16) ? 8 : 0)) * ldw + n0 + (lane & 15);
  v16h v;
#pragma unroll
  for (int i = 0; i < 8; ++i) {
    v[i]     = p[(size_t)i * ldw];
    v[8 + i] = p[(size_t)(16 + i) * ldw];
  }
  return v;
}

// One wave computes a 16x16 f32 tile: acc += A[m0:16, :] * W[:, n0:16]
__device__ __forceinline__ v8f wave_tile(const _Float16* A, int lda,
                                         const _Float16* W, int ldw,
                                         int m0, int n0, int Kdim, int lane) {
  v8f acc = {};
  for (int k0 = 0; k0 < Kdim; k0 += 32) {
    v16h a = load_fragA(A + m0 * lda, lda, k0, lane);
    v16h b = load_fragB(W, ldw, k0, n0, lane);
    acc = __builtin_amdgcn_wmma_f32_16x16x32_f16(false, a, false, b, (short)0, acc,
                                                 false, false);
  }
  return acc;
}

// GEMM layer with tanh epilogue, f16 output to LDS (padded stride ldh)
__device__ __forceinline__ void gemm_tanh(const _Float16* A, int lda,
                                          const _Float16* W,
                                          const float* __restrict__ bias,
                                          _Float16* H, int ldh,
                                          int Kdim, int N, int lane, int wave) {
  const int nTilesN = N >> 4;
  const int nTiles  = (BM >> 4) * nTilesN;
  for (int t = wave; t < nTiles; t += 8) {
    int m0 = (t / nTilesN) << 4;
    int n0 = (t % nTilesN) << 4;
    v8f acc = wave_tile(A, lda, W, N, m0, n0, Kdim, lane);
    int n   = n0 + (lane & 15);
    int mh  = m0 + ((lane & 16) ? 8 : 0);
    float bv = bias[n];
#pragma unroll
    for (int r = 0; r < 8; ++r)
      H[(mh + r) * ldh + n] = (_Float16)fast_tanh(acc[r] + bv);
  }
}

// ---------------------------------------------------------------------------
// Fused MLP kernel: both nets, final combine.
// LDS budget: 8704 + 16896*2 + 16896 = 59392 B  (< 64 KB)
// ---------------------------------------------------------------------------
#define XS_LD 136   // 128 + 8 halfs pad (272 B rows, 16B aligned)
#define H_LD  264   // 256 + 8 halfs pad (528 B rows, 16B aligned)
#define DR_LD 132   // floats

__global__ __launch_bounds__(256) void mlp_kernel(
    const float* __restrict__ x, const float* __restrict__ noise,
    const float* __restrict__ db1, const float* __restrict__ db2, const float* __restrict__ db3,
    const float* __restrict__ gb1, const float* __restrict__ gb2, const float* __restrict__ gb3,
    const _Float16* __restrict__ dw1c, const _Float16* __restrict__ dw2c, const _Float16* __restrict__ dw3c,
    const _Float16* __restrict__ gw1c, const _Float16* __restrict__ gw2c, const _Float16* __restrict__ gw3c,
    const float* __restrict__ fd, float* __restrict__ out) {
  __shared__ _Float16 xs[BM * XS_LD];
  __shared__ _Float16 hA[BM * H_LD];
  __shared__ _Float16 hB[BM * H_LD];
  __shared__ float    dr[BM * DR_LD];

  const int b0   = blockIdx.x * BM;
  const int lane = threadIdx.x & 31;
  const int wave = threadIdx.x >> 5;

  // stage x_current tile as f16 A-operand source
  for (int i = threadIdx.x; i < BM * D_DIM; i += 256) {
    int r = i >> 7, cidx = i & 127;
    xs[r * XS_LD + cidx] = (_Float16)x[(size_t)(b0 + r) * D_DIM + cidx];
  }
  __syncthreads();

  for (int net = 0; net < 2; ++net) {
    const _Float16* w1 = net ? gw1c : dw1c;
    const _Float16* w2 = net ? gw2c : dw2c;
    const _Float16* w3 = net ? gw3c : dw3c;
    const float* b1 = net ? gb1 : db1;
    const float* b2 = net ? gb2 : db2;
    const float* b3 = net ? gb3 : db3;

    gemm_tanh(xs, XS_LD, w1, b1, hA, H_LD, D_DIM, HID_DIM, lane, wave);
    __syncthreads();
    gemm_tanh(hA, H_LD, w2, b2, hB, H_LD, HID_DIM, HID_DIM, lane, wave);
    __syncthreads();

    // layer 3: linear head, N = 128
    const int nTilesN = D_DIM >> 4;                 // 8
    const int nTiles  = (BM >> 4) * nTilesN;        // 16
    for (int t = wave; t < nTiles; t += 8) {
      int m0 = (t / nTilesN) << 4;
      int n0 = (t % nTilesN) << 4;
      v8f acc = wave_tile(hB, H_LD, w3, D_DIM, m0, n0, HID_DIM, lane);
      int n  = n0 + (lane & 15);
      int mh = m0 + ((lane & 16) ? 8 : 0);
      float bv = b3[n];
      if (net == 0) {
#pragma unroll
        for (int r = 0; r < 8; ++r)
          dr[(mh + r) * DR_LD + n] = acc[r] + bv;    // drift (linear)
      } else {
#pragma unroll
        for (int r = 0; r < 8; ++r) {
          float t3  = acc[r] + bv;
          float dif = (t3 > 20.0f) ? t3 : __logf(1.0f + __expf(t3));   // softplus
          int b = b0 + mh + r;
          size_t idx = (size_t)b * D_DIM + n;
          float o = x[idx]
                  + dr[(mh + r) * DR_LD + n] * DT_F
                  + dif * (noise[b] * DT_POW_H_F)
                  + fd[idx] * ALPHA_DT_F;
          out[idx] = o;
        }
      }
    }
    __syncthreads();
  }
}

// ---------------------------------------------------------------------------
extern "C" void kernel_launch(void* const* d_in, const int* in_sizes, int n_in,
                              void* d_out, int out_size, void* d_ws, size_t ws_size,
                              hipStream_t stream) {
  const float* x_current = (const float*)d_in[0];
  const float* x_history = (const float*)d_in[1];
  const float* noise     = (const float*)d_in[2];
  const float* dw1 = (const float*)d_in[3];  const float* db1 = (const float*)d_in[4];
  const float* dw2 = (const float*)d_in[5];  const float* db2 = (const float*)d_in[6];
  const float* dw3 = (const float*)d_in[7];  const float* db3 = (const float*)d_in[8];
  const float* gw1 = (const float*)d_in[9];  const float* gb1 = (const float*)d_in[10];
  const float* gw2 = (const float*)d_in[11]; const float* gb2 = (const float*)d_in[12];
  const float* gw3 = (const float*)d_in[13]; const float* gb3 = (const float*)d_in[14];
  float* out = (float*)d_out;

  // workspace layout (bytes)
  char* ws = (char*)d_ws;
  float*     c    = (float*)(ws);                       // 1024 f           (4 KB)
  float*     fd   = (float*)(ws + 4096);                // B*D f            (1 MB)
  char* wbase     = ws + 4096 + (size_t)B_TOTAL * D_DIM * 4;
  _Float16* dw1c  = (_Float16*)(wbase);                         // 128*256 h
  _Float16* dw2c  = (_Float16*)(wbase + 65536);                 // 256*256 h
  _Float16* dw3c  = (_Float16*)(wbase + 65536 + 131072);        // 256*128 h
  _Float16* gw1c  = (_Float16*)(wbase + 65536 + 131072 + 65536);
  _Float16* gw2c  = (_Float16*)(wbase + 65536 + 131072 + 65536 + 65536);
  _Float16* gw3c  = (_Float16*)(wbase + 65536 + 131072 + 65536 + 65536 + 131072);

  // 1) coefficients
  coeff_kernel<<<(K_MEM + 255) / 256, 256, 0, stream>>>(c);

  // 2) weight convert to f16 (layout preserved: [K][N])
  wconv_kernel<<<(D_DIM * HID_DIM + 255) / 256, 256, 0, stream>>>(dw1, dw1c, D_DIM * HID_DIM);
  wconv_kernel<<<(HID_DIM * HID_DIM + 255) / 256, 256, 0, stream>>>(dw2, dw2c, HID_DIM * HID_DIM);
  wconv_kernel<<<(HID_DIM * D_DIM + 255) / 256, 256, 0, stream>>>(dw3, dw3c, HID_DIM * D_DIM);
  wconv_kernel<<<(D_DIM * HID_DIM + 255) / 256, 256, 0, stream>>>(gw1, gw1c, D_DIM * HID_DIM);
  wconv_kernel<<<(HID_DIM * HID_DIM + 255) / 256, 256, 0, stream>>>(gw2, gw2c, HID_DIM * HID_DIM);
  wconv_kernel<<<(HID_DIM * D_DIM + 255) / 256, 256, 0, stream>>>(gw3, gw3c, HID_DIM * D_DIM);

  // 3) HBM-bound single pass over x_history
  frac_kernel<<<B_TOTAL / 8, 256, 0, stream>>>(x_history, c, fd);

  // 4) WMMA MLPs + fused combine
  mlp_kernel<<<B_TOTAL / BM, 256, 0, stream>>>(
      x_current, noise, db1, db2, db3, gb1, gb2, gb3,
      dw1c, dw2c, dw3c, gw1c, gw2c, gw3c, fd, out);
}
